// TransformerMambaBlock_28690381537595
// MI455X (gfx1250) — matrix operand order repaired and checked by reading, not compiled, not run on hardware
//
#include <hip/hip_runtime.h>
#include <hip/hip_bf16.h>
#include <math.h>

// ---------------------------------------------------------------------------
// Mamba block for MI455X (gfx1250, wave32, WMMA), bf16 data path.
//   B=2, L=4096, D_MODEL=512, D_INNER=1024, D_HALF=512, DT_RANK=32, N_STATE=16
//
// All GEMM operands are materialized once as bf16 so the WMMA inner loop is
// pure global_load_b128 + v_wmma_f32_16x16x32_bf16 (no per-iteration cvt VALU),
// and GEMM memory traffic is halved (the workload is bandwidth-bound:
// ~18 GFLOP vs ~150 MB of traffic at 23.3 TB/s).
// ---------------------------------------------------------------------------

#define BATCH   2
#define SEQLEN  4096
#define DMODEL  512
#define DINNER  1024
#define DHALF   512
#define DTRANK  32
#define NSTATE  16
#define MROWS   (BATCH * SEQLEN)   // 8192

typedef __attribute__((ext_vector_type(16))) __bf16 v16bf;
typedef __attribute__((ext_vector_type(8)))  __bf16 v8bf;
typedef __attribute__((ext_vector_type(4)))  __bf16 v4bf;
typedef __attribute__((ext_vector_type(8)))  float  v8f;

__device__ __forceinline__ __bf16 f2bf(float f) {
  union { float f; unsigned u; } v; v.f = f;
  unsigned r = v.u + 0x7FFFu + ((v.u >> 16) & 1u);   // round-to-nearest-even
  return __builtin_bit_cast(__bf16, (unsigned short)(r >> 16));
}

__device__ __forceinline__ float softplus_f(float x) {
  return (x > 20.f) ? x : log1pf(__expf(x));
}

// ---------------------------------------------------------------------------
// Elementwise f32 -> bf16 (4 elements/thread; all sizes are multiples of 4).
// ---------------------------------------------------------------------------
__global__ __launch_bounds__(256)
void cvt_f32_to_bf16(const float* __restrict__ in, __bf16* __restrict__ out,
                     int n4) {
  const int i = blockIdx.x * 256 + threadIdx.x;
  if (i >= n4) return;
  const float4 v = ((const float4*)in)[i];
  v4bf o;
  o[0] = f2bf(v.x); o[1] = f2bf(v.y); o[2] = f2bf(v.z); o[3] = f2bf(v.w);
  ((v4bf*)out)[i] = o;
}

// ---------------------------------------------------------------------------
// WMMA GEMM:  C[M,N] = A[M,K] * W[N,K]^T   (bf16 operands, f32 accumulate)
//   block = 128 threads = 4 waves; block tile 64Mx64N; wave tile 32x32.
//   EPI 0: store f32 C
//   EPI 1: store f32 softplus(acc + bias[n])
//   EPI 2: store f32 C and bf16 Cbf
// Fragment layouts per CDNA5 ISA 7.12.2:
//   A 16x32: lane r(0..15) row=r, K={0..7,16..23}; lanes 16..31 K={8..15,24..31}
//   B 32x16: col n=lane&15, K same as A: {0..7,16..23} / {8..15,24..31} of W row
//   C/D    : N = lane&15, M = vgpr_j + 8*(lane>>4)
// ---------------------------------------------------------------------------
template <int EPI>
__global__ __launch_bounds__(128)
void gemm_bf16_wmma(const __bf16* __restrict__ A, int lda,
                    const __bf16* __restrict__ W,        // (N,K) row-major
                    float* __restrict__ C, int ldc,
                    int K,
                    const float* __restrict__ bias,
                    __bf16* __restrict__ Cbf) {
  const int lane = threadIdx.x & 31;
  const int wave = threadIdx.x >> 5;        // 0..3
  const int m0   = blockIdx.x * 64 + (wave >> 1) * 32;
  const int n0   = blockIdx.y * 64 + (wave & 1) * 32;
  const int r    = lane & 15;
  const int kh   = lane >> 4;               // 0 or 1

  v8f acc[2][2] = {};

  for (int k0 = 0; k0 < K; k0 += 32) {
    if (k0 + 32 < K) {                      // gfx1250 global_prefetch_b8
      __builtin_prefetch(A + (size_t)(m0 + r) * lda + k0 + 32, 0, 1);
      __builtin_prefetch(W + (size_t)(n0 + r) * K + k0 + 32, 0, 1);
    }
    v16bf afrag[2], bfrag[2];
#pragma unroll
    for (int t = 0; t < 2; ++t) {
      // A: two 8-bf16 (16B) runs per lane -> 2x global_load_b128
      const __bf16* ap = A + (size_t)(m0 + t * 16 + r) * lda + k0 + kh * 8;
      v8bf a0 = *(const v8bf*)(ap + 0);
      v8bf a1 = *(const v8bf*)(ap + 16);
      afrag[t] = __builtin_shufflevector(a0, a1, 0, 1, 2, 3, 4, 5, 6, 7,
                                         8, 9, 10, 11, 12, 13, 14, 15);
      // B: two 8-bf16 (16B) runs per lane (same K order as A) -> 2x global_load_b128
      const __bf16* wp = W + (size_t)(n0 + t * 16 + r) * K + k0 + kh * 8;
      v8bf b0 = *(const v8bf*)(wp + 0);
      v8bf b1 = *(const v8bf*)(wp + 16);
      bfrag[t] = __builtin_shufflevector(b0, b1, 0, 1, 2, 3, 4, 5, 6, 7,
                                         8, 9, 10, 11, 12, 13, 14, 15);
    }
#pragma unroll
    for (int tm = 0; tm < 2; ++tm)
#pragma unroll
      for (int tn = 0; tn < 2; ++tn)
        acc[tm][tn] = __builtin_amdgcn_wmma_f32_16x16x32_bf16(
            false, afrag[tm], false, bfrag[tn],
            (short)0, acc[tm][tn], false, false);
  }

#pragma unroll
  for (int tm = 0; tm < 2; ++tm)
#pragma unroll
    for (int tn = 0; tn < 2; ++tn) {
      const int n = n0 + tn * 16 + r;
#pragma unroll
      for (int j = 0; j < 8; ++j) {
        const int m = m0 + tm * 16 + j + 8 * kh;
        float v = acc[tm][tn][j];
        if (EPI == 1) v = softplus_f(v + bias[n]);
        C[(size_t)m * ldc + n] = v;
        if (EPI == 2) Cbf[(size_t)m * ldc + n] = f2bf(v);
      }
    }
}

// ---------------------------------------------------------------------------
// Depthwise conv1d (K=4, pad left=1 right=2) + SiLU, dual-precision output.
//   out_f  (optional): f32 at column (ocol+c) with leading dim ldo
//   out_bf (optional): bf16 at column (ocol+c) with leading dim ldo
// ---------------------------------------------------------------------------
__global__ __launch_bounds__(256)
void dwconv_silu(const float* __restrict__ in, int ldi,
                 const float* __restrict__ w,   // (C,1,4)
                 float* __restrict__ out_f,
                 __bf16* __restrict__ out_bf,
                 int ldo, int ocol) {
  const int idx = blockIdx.x * blockDim.x + threadIdx.x;
  const int C   = DHALF;
  if (idx >= BATCH * SEQLEN * C) return;
  const int c = idx % C;
  const int l = (idx / C) % SEQLEN;
  const int b = idx / (C * SEQLEN);

  const float* col = in + (size_t)b * SEQLEN * ldi + c;
  float s = 0.f;
#pragma unroll
  for (int j = 0; j < 4; ++j) {
    const int ll = l - 1 + j;                // left pad 1, right pad 2
    if (ll >= 0 && ll < SEQLEN) s += col[(size_t)ll * ldi] * w[c * 4 + j];
  }
  const float y = s / (1.f + __expf(-s));    // SiLU
  const size_t o = ((size_t)b * SEQLEN + l) * ldo + ocol + c;
  if (out_f)  out_f[o]  = y;
  if (out_bf) out_bf[o] = f2bf(y);
}

// ---------------------------------------------------------------------------
// Selective scan. 4 blocks x 256 threads: thread <-> (b, d); h[16] in regs.
// Per timestep, B/C (32 floats) staged via double-buffered LDS, 1 barrier.
// Writes y (bf16) into ycat_bf[:, 0:512]; z half already written by conv.
// ---------------------------------------------------------------------------
__global__ __launch_bounds__(256)
void selective_scan(const float* __restrict__ u,      // xconv f32 (B,L,512)
                    const float* __restrict__ delta,  // f32 (B,L,512)
                    const float* __restrict__ A_log,  // (512,16)
                    const float* __restrict__ xdbl,   // f32 (B,L,64): [dt|B|C]
                    const float* __restrict__ Dp,     // (512)
                    __bf16* __restrict__ ycat_bf) {   // bf16 (B,L,1024)
  const int b = blockIdx.x >> 1;
  const int d = (blockIdx.x & 1) * 256 + threadIdx.x;

  __shared__ float Bsh[2][NSTATE];
  __shared__ float Csh[2][NSTATE];

  float A_row[NSTATE], h[NSTATE];
#pragma unroll
  for (int n = 0; n < NSTATE; ++n) {
    A_row[n] = -__expf(A_log[d * NSTATE + n]);
    h[n] = 0.f;
  }
  const float Dd = Dp[d];

  for (int l = 0; l < SEQLEN; ++l) {
    const int buf = l & 1;
    if (threadIdx.x < 32) {
      const float v = xdbl[((size_t)b * SEQLEN + l) * 64 + DTRANK + threadIdx.x];
      if (threadIdx.x < 16) Bsh[buf][threadIdx.x] = v;
      else                  Csh[buf][threadIdx.x - 16] = v;
    }
    __syncthreads();

    const size_t off = ((size_t)b * SEQLEN + l) * DHALF + d;
    const float dl = delta[off];
    const float uv = u[off];
    const float dlu = dl * uv;
    float y = 0.f;
#pragma unroll
    for (int n = 0; n < NSTATE; ++n) {
      h[n] = __expf(dl * A_row[n]) * h[n] + dlu * Bsh[buf][n];
      y += h[n] * Csh[buf][n];
    }
    ycat_bf[((size_t)b * SEQLEN + l) * DINNER + d] = f2bf(y + uv * Dd);
  }
}

// ---------------------------------------------------------------------------
// Launch
// ---------------------------------------------------------------------------
extern "C" void kernel_launch(void* const* d_in, const int* in_sizes, int n_in,
                              void* d_out, int out_size, void* d_ws, size_t ws_size,
                              hipStream_t stream) {
  const float* hidden     = (const float*)d_in[0];  // (2,4096,512)
  const float* in_proj_w  = (const float*)d_in[1];  // (1024,512)
  const float* conv_x_w   = (const float*)d_in[2];  // (512,1,4)
  const float* conv_z_w   = (const float*)d_in[3];  // (512,1,4)
  const float* x_proj_w   = (const float*)d_in[4];  // (64,512)
  const float* dt_proj_w  = (const float*)d_in[5];  // (512,32)
  const float* dt_proj_b  = (const float*)d_in[6];  // (512)
  const float* A_log      = (const float*)d_in[7];  // (512,16)
  const float* D_param    = (const float*)d_in[8];  // (512)
  const float* out_proj_w = (const float*)d_in[9];  // (512,1024)
  float* out = (float*)d_out;                       // (2,4096,512) f32

  // ---- workspace partition -------------------------------------------------
  char* p = (char*)d_ws;
  auto alloc_f = [&](size_t n) { float*  q = (float*)p;  p += n * 4; return q; };
  auto alloc_b = [&](size_t n) { __bf16* q = (__bf16*)p; p += n * 2; return q; };

  float*  xz_f     = alloc_f((size_t)MROWS * DINNER);   // in_proj out (f32, conv in)
  float*  xconv_f  = alloc_f((size_t)MROWS * DHALF);    // conv-x out f32 (scan u)
  float*  xdbl_f   = alloc_f((size_t)MROWS * 64);       // x_proj out f32 (scan B/C)
  float*  delta_f  = alloc_f((size_t)MROWS * DHALF);    // dt_proj out f32 (scan)
  __bf16* hidden_b = alloc_b((size_t)MROWS * DMODEL);   // bf16 GEMM operands:
  __bf16* xconv_b  = alloc_b((size_t)MROWS * DHALF);
  __bf16* xdbl_b   = alloc_b((size_t)MROWS * 64);
  __bf16* ycat_b   = alloc_b((size_t)MROWS * DINNER);   // [y | z_conv] for out_proj
  __bf16* w_in_b   = alloc_b((size_t)DINNER * DMODEL);
  __bf16* w_xp_b   = alloc_b((size_t)(DTRANK + 2 * NSTATE) * DHALF);
  __bf16* w_dt_b   = alloc_b((size_t)DHALF * DTRANK);
  __bf16* w_out_b  = alloc_b((size_t)DMODEL * DINNER);

  const dim3 blk(128);
  auto cvt = [&](const float* src, __bf16* dst, size_t n) {
    const int n4 = (int)(n / 4);
    cvt_f32_to_bf16<<<(n4 + 255) / 256, 256, 0, stream>>>(src, dst, n4);
  };

  // ---- 0) one-time bf16 materialization -----------------------------------
  cvt(hidden,     hidden_b, (size_t)MROWS * DMODEL);
  cvt(in_proj_w,  w_in_b,   (size_t)DINNER * DMODEL);
  cvt(x_proj_w,   w_xp_b,   (size_t)(DTRANK + 2 * NSTATE) * DHALF);
  cvt(dt_proj_w,  w_dt_b,   (size_t)DHALF * DTRANK);
  cvt(out_proj_w, w_out_b,  (size_t)DMODEL * DINNER);

  // ---- 1) in_proj: (8192,512) x (1024,512)^T -> xz_f ----------------------
  gemm_bf16_wmma<0><<<dim3(MROWS / 64, DINNER / 64), blk, 0, stream>>>(
      hidden_b, DMODEL, w_in_b, xz_f, DINNER, DMODEL, nullptr, nullptr);

  // ---- 2/3) depthwise conv + SiLU -----------------------------------------
  const int convN = BATCH * SEQLEN * DHALF;
  dwconv_silu<<<(convN + 255) / 256, 256, 0, stream>>>(
      xz_f, DINNER, conv_x_w, xconv_f, xconv_b, DHALF, 0);
  dwconv_silu<<<(convN + 255) / 256, 256, 0, stream>>>(
      xz_f + DHALF, DINNER, conv_z_w, nullptr, ycat_b, DINNER, DHALF);

  // ---- 4) x_proj: (8192,512) x (64,512)^T -> xdbl (f32 + bf16) ------------
  gemm_bf16_wmma<2><<<dim3(MROWS / 64, 64 / 64), blk, 0, stream>>>(
      xconv_b, DHALF, w_xp_b, xdbl_f, 64, DHALF, nullptr, xdbl_b);

  // ---- 5) dt_proj + bias + softplus: (8192,32) x (512,32)^T -> delta ------
  gemm_bf16_wmma<1><<<dim3(MROWS / 64, DHALF / 64), blk, 0, stream>>>(
      xdbl_b, 64, w_dt_b, delta_f, DHALF, DTRANK, dt_proj_b, nullptr);

  // ---- 6) selective scan -> ycat_b[:, 0:512] ------------------------------
  selective_scan<<<BATCH * (DHALF / 256), 256, 0, stream>>>(
      xconv_f, delta_f, A_log, xdbl_f, D_param, ycat_b);

  // ---- 7) out_proj: (8192,1024) x (512,1024)^T -> out ---------------------
  gemm_bf16_wmma<0><<<dim3(MROWS / 64, DHALF / 64), blk, 0, stream>>>(
      ycat_b, DINNER, w_out_b, out, DHALF, DINNER, nullptr, nullptr);
}
